// AttentionBlock_5669356831785
// MI455X (gfx1250) — compile-verified, matrix operand order checked by reading, not run on hardware
//
#include <hip/hip_runtime.h>

typedef __attribute__((ext_vector_type(16))) _Float16 v16h;
typedef __attribute__((ext_vector_type(8)))  float    v8f;

constexpr int NB = 4;
constexpr int NT = 4096;
constexpr int NC = 256;   // in channels
constexpr int NK = 256;   // key size
constexpr int NV = 256;   // value size
constexpr float INV_SQRT_K = 0.0625f;   // 1/sqrt(256)
constexpr size_t QKV_ELEMS = (size_t)NB * NT * NK;

__device__ __forceinline__ v8f wmma16(v16h a, v16h b, v8f c) {
  // D = A(16x32 f16) * B(32x16 f16) + C(16x16 f32)
  return __builtin_amdgcn_wmma_f32_16x16x32_f16(false, a, false, b,
                                                (short)0, c, false, false);
}

// CDNA5 async copy: per-lane 16B global -> LDS, tracked by ASYNCcnt.
__device__ __forceinline__ void async_copy_b128(uint32_t lds_addr,
                                                const void* gaddr) {
  asm volatile("global_load_async_to_lds_b128 %0, %1, off"
               :: "v"(lds_addr), "v"(gaddr) : "memory");
}
__device__ __forceinline__ void wait_async0() {
  asm volatile("s_wait_asynccnt 0x0" ::: "memory");
}
__device__ __forceinline__ uint32_t lds_off(const void* p) {
  return (uint32_t)(size_t)p;   // low 32 bits of flat LDS address = LDS offset
}

// ---------------------------------------------------------------------------
// Kernel 1: QKV projection.  q[b,t,k] = sum_c x[b,c,t] * W[k,c] + bias[k]
// q,k stored [B,T,K] f16; v stored TRANSPOSED [B,V,T] f16 (feeds out_gemm
// B-fragments as contiguous rows -> async-copyable).
// grid: (T/16, K/64, B*3), block 128 (4 waves, one 16x16 k-tile per wave)
// ---------------------------------------------------------------------------
__global__ __launch_bounds__(128) void qkv_proj(
    const float* __restrict__ x,
    const float* __restrict__ Wq, const float* __restrict__ bq,
    const float* __restrict__ Wk, const float* __restrict__ bk,
    const float* __restrict__ Wv, const float* __restrict__ bv,
    _Float16* __restrict__ qh, _Float16* __restrict__ kh,
    _Float16* __restrict__ vT)
{
  __shared__ __align__(32) _Float16 xs[16 * NC];   // x^T tile, f16
  const int tid = threadIdx.x;
  const int t0 = blockIdx.x * 16;
  const int z = blockIdx.z;
  const int b = z / 3, which = z % 3;
  const float* W    = (which == 0) ? Wq : (which == 1) ? Wk : Wv;
  const float* bias = (which == 0) ? bq : (which == 1) ? bk : bv;

  // stage x[b, :, t0..t0+15] transposed + converted to f16 (coalesced over t)
  for (int i = tid; i < 16 * NC; i += 128) {
    int m = i & 15, c = i >> 4;
    xs[m * NC + c] = (_Float16)x[((size_t)b * NC + c) * NT + t0 + m];
  }
  __syncthreads();

  const int wave = tid >> 5, lane = tid & 31;
  const int half = lane >> 4, lm = lane & 15;
  const int kbase = blockIdx.y * 64 + wave * 16;

  v8f acc = {};
  #pragma unroll
  for (int kc = 0; kc < NC / 32; ++kc) {
    v16h a = *(const v16h*)&xs[lm * NC + kc * 32 + half * 16];
    v16h bf;
    const float* wr = W + (size_t)(kbase + lm) * NC + kc * 32 + half * 16;
    #pragma unroll
    for (int i = 0; i < 16; ++i) bf[i] = (_Float16)wr[i];
    acc = wmma16(a, bf, acc);
  }
  const float bn = bias[kbase + lm];
  #pragma unroll
  for (int r = 0; r < 8; ++r) {
    int t = t0 + r + 8 * half;
    _Float16 hv = (_Float16)(acc[r] + bn);
    if (which == 2)       // v transposed: vT[b][vcol][t]
      vT[((size_t)b * NV + kbase + lm) * NT + t] = hv;
    else if (which == 1)
      kh[((size_t)b * NT + t) * NK + kbase + lm] = hv;
    else
      qh[((size_t)b * NT + t) * NK + kbase + lm] = hv;
  }
}

// ---------------------------------------------------------------------------
// Kernel 2: scores = (Q K^T)/sqrt(K), causal mask (s > t -> -inf).
// Fully-masked 32x64 blocks are SKIPPED (no writes; k3/k4 never read them).
// grid: (T/32, T/64, B), block 256 (8 waves, 2x4 arrangement of 16x16 tiles)
// ---------------------------------------------------------------------------
__global__ __launch_bounds__(256) void scores_kernel(
    const _Float16* __restrict__ qh, const _Float16* __restrict__ kh,
    float* __restrict__ wts)
{
  const float NEG_INF = -__builtin_inff();
  const int tid = threadIdx.x;
  const int t0 = blockIdx.x * 32;
  const int s0 = blockIdx.y * 64;
  const int b  = blockIdx.z;

  if (s0 > t0 + 31) return;   // whole block above diagonal: nothing to do

  float* wb = wts + (size_t)b * NT * NT;

  __shared__ __align__(32) _Float16 qs[32 * NK];
  __shared__ __align__(32) _Float16 ks[64 * NK];
  {
    const _Float16* qsrc = qh + ((size_t)b * NT + t0) * NK;   // 32 rows
    #pragma unroll
    for (int k = 0; k < 4; ++k) {
      int idx = tid + k * 256;                // 1024 x 16B = 16KB
      async_copy_b128(lds_off(&qs[idx * 8]), qsrc + idx * 8);
    }
    const _Float16* ksrc = kh + ((size_t)b * NT + s0) * NK;   // 64 rows
    #pragma unroll
    for (int k = 0; k < 8; ++k) {
      int idx = tid + k * 256;                // 2048 x 16B = 32KB
      async_copy_b128(lds_off(&ks[idx * 8]), ksrc + idx * 8);
    }
  }
  wait_async0();
  __syncthreads();

  const int wave = tid >> 5, lane = tid & 31;
  const int half = lane >> 4, lm = lane & 15;
  const int wt = wave >> 2, wsv = wave & 3;
  const int tw = t0 + wt * 16, sw = s0 + wsv * 16;

  if (sw > tw + 15) {  // wave tile fully masked (still inside a kept block)
    #pragma unroll
    for (int r = 0; r < 8; ++r)
      wb[(size_t)(tw + r + 8 * half) * NT + sw + lm] = NEG_INF;
    return;
  }

  v8f acc = {};
  #pragma unroll
  for (int kc = 0; kc < NK / 32; ++kc) {
    v16h a  = *(const v16h*)&qs[(wt * 16 + lm) * NK + kc * 32 + half * 16];
    v16h bf = *(const v16h*)&ks[(wsv * 16 + lm) * NK + kc * 32 + half * 16];
    acc = wmma16(a, bf, acc);
  }
  #pragma unroll
  for (int r = 0; r < 8; ++r) {
    int t = tw + r + 8 * half, s = sw + lm;
    wb[(size_t)t * NT + s] = (s > t) ? NEG_INF : acc[r] * INV_SQRT_K;
  }
}

// ---------------------------------------------------------------------------
// Kernel 3: column softmax stats (softmax over axis=1 == the query/t axis).
// One thread per column s; online max/sum; reads only the valid t >= s region.
// grid: B*T/256, block 256
// ---------------------------------------------------------------------------
__global__ __launch_bounds__(256) void col_stats(
    const float* __restrict__ wts, float* __restrict__ cmax,
    float* __restrict__ crsum)
{
  const int gid = blockIdx.x * 256 + threadIdx.x;
  const int b = gid >> 12, s = gid & (NT - 1);
  const float* col = wts + (size_t)b * NT * NT + s;
  float m = -__builtin_inff(), sum = 0.f;
  for (int t = s & ~255; t < NT; ++t) {   // uniform start; predicate t >= s
    if (t >= s) {
      float xv = col[(size_t)t * NT];
      __builtin_prefetch(&col[(size_t)(t + 16) * NT], 0, 0);
      float nm = fmaxf(m, xv);
      sum = sum * __expf(m - nm) + __expf(xv - nm);
      m = nm;
    }
  }
  cmax[gid]  = m;
  crsum[gid] = 1.0f / sum;
}

// ---------------------------------------------------------------------------
// Kernel 4: weights = exp(x - colmax)/colsum.  Elements in k2-skipped blocks
// are written as 0 WITHOUT reading (they were never written). float4 pass.
// ---------------------------------------------------------------------------
__global__ __launch_bounds__(256) void normalize_kernel(
    float* __restrict__ wts, const float* __restrict__ cmax,
    const float* __restrict__ crsum)
{
  const size_t e = ((size_t)blockIdx.x * 256 + threadIdx.x) * 4;
  const int b = (int)(e >> 24);                   // / (T*T) = 2^24
  const int t = (int)((e >> 12) & (size_t)(NT - 1));
  const int s = (int)(e & (size_t)(NT - 1));

  if ((s & ~63) > (t & ~31) + 31) {               // k2-skipped block -> zeros
    float4 z = make_float4(0.f, 0.f, 0.f, 0.f);
    *(float4*)(wts + e) = z;
    return;
  }
  const float NEG_INF = -__builtin_inff();
  float4 v = *(float4*)(wts + e);
  const float* m  = cmax  + b * NT + s;
  const float* rs = crsum + b * NT + s;
  v.x = (v.x > NEG_INF) ? __expf(v.x - m[0]) * rs[0] : 0.f;
  v.y = (v.y > NEG_INF) ? __expf(v.y - m[1]) * rs[1] : 0.f;
  v.z = (v.z > NEG_INF) ? __expf(v.z - m[2]) * rs[2] : 0.f;
  v.w = (v.w > NEG_INF) ? __expf(v.w - m[3]) * rs[3] : 0.f;
  *(float4*)(wts + e) = v;
}

// ---------------------------------------------------------------------------
// Kernel 5: out[b,v,t] = sum_s weights[b,t,s] * v[b,s,v]  (transposed store).
// grid: (T/32, B), block 256; each wave owns 16t x 64v (4 accumulators).
// Double-buffered LDS; V tiles fetched with async-to-LDS; K-loop truncated
// at the causal boundary (weights above the diagonal are zero).
// ---------------------------------------------------------------------------
__global__ __launch_bounds__(256) void out_gemm(
    const float* __restrict__ wts, const _Float16* __restrict__ vT,
    float* __restrict__ out)
{
  __shared__ __align__(32) _Float16 lw[2][32 * 32];   // weight tiles (f16)
  __shared__ __align__(32) _Float16 lv[2][NV * 32];   // V tiles [v][s]
  const int tid = threadIdx.x;
  const int t0 = blockIdx.x * 32;
  const int b  = blockIdx.y;
  const int wave = tid >> 5, lane = tid & 31;
  const int half = lane >> 4, lm = lane & 15;
  const int wt = wave >> 2, wv = wave & 3;
  const int kcend = (t0 >> 5) + 1;       // s <= t0+31 only (causal)

  const int wrow = tid >> 3, wc4 = (tid & 7) * 4;
  const float* wsrc =
      wts + (size_t)b * NT * NT + (size_t)(t0 + wrow) * NT + wc4;
  const _Float16* vbase = vT + (size_t)b * NV * NT;

  v8f acc[4] = {};

  // ---- prologue: stage tile 0 into buffer 0 ----
  {
    float4 w4 = *(const float4*)(wsrc);
    lw[0][wrow * 32 + wc4 + 0] = (_Float16)w4.x;
    lw[0][wrow * 32 + wc4 + 1] = (_Float16)w4.y;
    lw[0][wrow * 32 + wc4 + 2] = (_Float16)w4.z;
    lw[0][wrow * 32 + wc4 + 3] = (_Float16)w4.w;
    #pragma unroll
    for (int k = 0; k < 4; ++k) {
      int idx = tid + k * 256;           // 1024 x 16B = 16KB
      int row = idx >> 2, ch = (idx & 3) * 8;
      async_copy_b128(lds_off(&lv[0][row * 32 + ch]),
                      vbase + (size_t)row * NT + ch);
    }
  }
  wait_async0();
  __syncthreads();

  for (int kc = 0; kc < kcend; ++kc) {
    const int cur = kc & 1, nxt = cur ^ 1;
    const bool more = (kc + 1) < kcend;
    float4 w4;
    if (more) {
      w4 = *(const float4*)(wsrc + (kc + 1) * 32);       // overlaps compute
      #pragma unroll
      for (int k = 0; k < 4; ++k) {
        int idx = tid + k * 256;
        int row = idx >> 2, ch = (idx & 3) * 8;
        async_copy_b128(lds_off(&lv[nxt][row * 32 + ch]),
                        vbase + (size_t)row * NT + (kc + 1) * 32 + ch);
      }
    }

    v16h a = *(const v16h*)&lw[cur][(wt * 16 + lm) * 32 + half * 16];
    #pragma unroll
    for (int j = 0; j < 4; ++j) {
      v16h bf =
          *(const v16h*)&lv[cur][(wv * 64 + j * 16 + lm) * 32 + half * 16];
      acc[j] = wmma16(a, bf, acc[j]);
    }

    if (more) {
      lw[nxt][wrow * 32 + wc4 + 0] = (_Float16)w4.x;
      lw[nxt][wrow * 32 + wc4 + 1] = (_Float16)w4.y;
      lw[nxt][wrow * 32 + wc4 + 2] = (_Float16)w4.z;
      lw[nxt][wrow * 32 + wc4 + 3] = (_Float16)w4.w;
      wait_async0();
    }
    __syncthreads();
  }

  #pragma unroll
  for (int j = 0; j < 4; ++j) {
    #pragma unroll
    for (int r = 0; r < 8; ++r) {
      int t  = t0 + wt * 16 + r + 8 * half;
      int vc = wv * 64 + j * 16 + lm;
      out[((size_t)b * NV + vc) * NT + t] = acc[j][r];
    }
  }
}

// ---------------------------------------------------------------------------
extern "C" void kernel_launch(void* const* d_in, const int* in_sizes, int n_in,
                              void* d_out, int out_size, void* d_ws,
                              size_t ws_size, hipStream_t stream)
{
  (void)in_sizes; (void)n_in; (void)out_size; (void)ws_size;
  const float* x  = (const float*)d_in[0];
  const float* Wq = (const float*)d_in[1];
  const float* bq = (const float*)d_in[2];
  const float* Wk = (const float*)d_in[3];
  const float* bk = (const float*)d_in[4];
  const float* Wv = (const float*)d_in[5];
  const float* bv = (const float*)d_in[6];

  float* out = (float*)d_out;                       // [B,V,T]
  float* wts = out + (size_t)NB * NV * NT;          // [B,T,T]

  _Float16* qh = (_Float16*)d_ws;                   // f16 q  [B,T,K]
  _Float16* kh = qh + QKV_ELEMS;                    // f16 k  [B,T,K]
  _Float16* vT = kh + QKV_ELEMS;                    // f16 v  [B,V,T] (transposed)
  float* cmax  = (float*)(vT + QKV_ELEMS);          // [B*T]
  float* crsum = cmax + NB * NT;                    // [B*T]

  qkv_proj<<<dim3(NT / 16, NK / 64, NB * 3), 128, 0, stream>>>(
      x, Wq, bq, Wk, bk, Wv, bv, qh, kh, vT);

  scores_kernel<<<dim3(NT / 32, NT / 64, NB), 256, 0, stream>>>(qh, kh, wts);

  col_stats<<<dim3(NB * NT / 256), 256, 0, stream>>>(wts, cmax, crsum);

  normalize_kernel<<<dim3((unsigned)((size_t)NB * NT * NT / 4 / 256)), 256, 0,
                     stream>>>(wts, cmax, crsum);

  out_gemm<<<dim3(NT / 32, NB), 256, 0, stream>>>(wts, vT, out);
}
